// RNN_63668595196411
// MI455X (gfx1250) — compile-verified, matrix operand order checked by reading, not run on hardware
//
#include <hip/hip_runtime.h>
#include <hip/hip_bf16.h>

// ---------------- problem constants ----------------
#define B_    64
#define T_    2048
#define DIN_  256
#define H_    512
#define DOUT_ 128

#define NWG     16            // persistent workgroups (1 per WGP)
#define NCOL    (H_ / NWG)    // 32 output columns per WG per layer
#define THREADS 256           // 8 wave32 -> 8 (16x16) C tiles = 64x32 slice

// padded LDS strides in halves (16B row padding -> fragment rows spread banks)
#define XK (DIN_ + 8)         // 264
#define HK (H_  + 8)          // 520

// LDS layout (halves):
//  xbuf[2] : 2*64*XK     hbuf0/hbuf1 : 64*HK each
//  wih0 : 32*XK (tail-reused for W_out slice 16*HK)   whh0/wih1/whh1 : 32*HK
#define SMEM_HALVES (2*64*XK + 2*64*HK + 32*XK + 3*32*HK)
#define SMEM_BYTES  (SMEM_HALVES * 2)   // 317440 B  (<= 320KB WGP LDS)

#define USE_TDM 1

typedef _Float16 v16h __attribute__((ext_vector_type(16)));
typedef _Float16 v8h  __attribute__((ext_vector_type(8)));
typedef float    v8f  __attribute__((ext_vector_type(8)));
typedef int      si4  __attribute__((ext_vector_type(4)));
typedef int      si8  __attribute__((ext_vector_type(8)));

// ---- LDS byte offset of a generic pointer (addrspacecast -> ptrtoint) ------
__device__ __forceinline__ unsigned lds_off(const void* p) {
  return (unsigned)(unsigned long long)
         (__attribute__((address_space(3))) const void*)p;
}

// ---- Tensor Data Mover: 2D f16 tile global->LDS with per-row padding -------
// rows x width f16 elements, global row stride = stride elems, LDS row stride
// = width*2 + 16 bytes (pad_interval=256 DWORDs, pad_amount=4 DWORDs).
// Issued once per calling wave; tracked by TENSORcnt.  (ISA 08_async_tensor §8)
__device__ __forceinline__ void tdm_load_2d_f16(const void* gsrc, unsigned ldsbyte,
                                                int width, int rows, int stride) {
  unsigned long long ga = (unsigned long long)gsrc;
  si4 g0;
  g0[0] = 1;                                   // count=1, user mode
  g0[1] = (int)ldsbyte;                        // lds_addr (bytes)
  g0[2] = (int)(unsigned)(ga & 0xffffffffu);   // global_addr[31:0]
  g0[3] = (int)(((unsigned)(ga >> 32) & 0x01ffffffu) | (2u << 30)); // type=2
  si8 g1;
  g1[0] = (1 << 16)            // data_size = 2 bytes
        | (1 << 20)            // pad_enable
        | (7 << 22)            // pad_interval: 256 DWORDs (=1024B row)
        | (3 << 25);           // pad_amount  : 4 DWORDs   (=16B)
  g1[1] = (width & 0xffff) << 16;                       // tensor_dim0 lo16
  g1[2] = ((width >> 16) & 0xffff) | ((rows & 0xffff) << 16); // dim0 hi | dim1 lo
  g1[3] = ((rows >> 16) & 0xffff) | ((width & 0xffff) << 16); // dim1 hi | tile_dim0
  g1[4] = rows & 0xffff;                                // tile_dim1 (tile_dim2=0)
  g1[5] = stride;                                       // tensor_dim0_stride lo32
  g1[6] = 0;
  g1[7] = 0;
  si4 g2 = {1, 0, 0, 0};   // tensor_dim2=1 (tile dims 3/4 unused)
  si4 g3 = {0, 0, 0, 0};
  asm volatile("tensor_load_to_lds %0, %1, %2, %3"
               :: "s"(g0), "s"(g1), "s"(g2), "s"(g3)
               : "memory");
}

// ---- WMMA fragment load from LDS (ISA 7.12.2 16-bit A/B layout) ------------
__device__ __forceinline__ v16h load_frag(const _Float16* base, int stride,
                                          int row, int kb, int lane) {
  const _Float16* p = base + row * stride + kb + ((lane & 16) ? 8 : 0);
  v8h lo = *(const v8h*)(p);
  v8h hi = *(const v8h*)(p + 16);
  v16h r;
#pragma unroll
  for (int i = 0; i < 8; ++i) { r[i] = lo[i]; r[i + 8] = hi[i]; }
  return r;
}

__device__ __forceinline__ v8f gemm_acc(v8f acc,
                                        const _Float16* A, int as, int arow,
                                        const _Float16* Bm, int bs, int brow,
                                        int K, int lane) {
#pragma unroll
  for (int kb = 0; kb < K; kb += 32) {
    v16h a = load_frag(A, as, arow, kb, lane);
    v16h b = load_frag(Bm, bs, brow, kb, lane);
    acc = __builtin_amdgcn_wmma_f32_16x16x32_f16(false, a, false, b,
                                                 (short)0, acc, false, false);
  }
  return acc;
}

// ---- device-wide barrier for NWG persistent workgroups ----------------------
__device__ __forceinline__ void grid_barrier(unsigned* cnt, unsigned* gen,
                                             unsigned expect) {
  __syncthreads();
  __threadfence();
  if (threadIdx.x == 0) {
    if (__hip_atomic_fetch_add(cnt, 1u, __ATOMIC_ACQ_REL,
                               __HIP_MEMORY_SCOPE_AGENT) == NWG - 1) {
      __hip_atomic_store(cnt, 0u, __ATOMIC_RELAXED, __HIP_MEMORY_SCOPE_AGENT);
      __hip_atomic_fetch_add(gen, 1u, __ATOMIC_RELEASE, __HIP_MEMORY_SCOPE_AGENT);
    } else {
      while (__hip_atomic_load(gen, __ATOMIC_ACQUIRE,
                               __HIP_MEMORY_SCOPE_AGENT) <= expect) {
        __builtin_amdgcn_s_sleep(1);
      }
    }
  }
  __syncthreads();
  __threadfence();
}

// ---- init: reset barrier state, seed h exchange (graph-replay safe) --------
__global__ void rnn_init(const float* __restrict__ h0in,
                         _Float16* __restrict__ h0g, _Float16* __restrict__ h1g,
                         unsigned* __restrict__ cnt, unsigned* __restrict__ gen) {
  int i = blockIdx.x * blockDim.x + threadIdx.x;
  int stride = gridDim.x * blockDim.x;
  if (i == 0) { *cnt = 0u; *gen = 0u; }
  for (int k = i; k < 2 * B_ * H_; k += stride) h0g[k] = (_Float16)0.f;
  for (int k = i; k < B_ * H_; k += stride) {
    h1g[k] = (_Float16)0.f;                            // parity-0 buffer
    h1g[B_ * H_ + k] = (_Float16)h0in[B_ * H_ + k];    // parity-1 = initial h1
  }
}

// ---- persistent scan kernel -------------------------------------------------
__global__ __launch_bounds__(THREADS)
void rnn_scan(const float* __restrict__ x,    const float* __restrict__ h0in,
              const float* __restrict__ Wih0, const float* __restrict__ bih0,
              const float* __restrict__ Whh0, const float* __restrict__ bhh0,
              const float* __restrict__ Wih1, const float* __restrict__ bih1,
              const float* __restrict__ Whh1, const float* __restrict__ bhh1,
              const float* __restrict__ Wout, const float* __restrict__ bout,
              float* __restrict__ out,
              _Float16* __restrict__ h0g, _Float16* __restrict__ h1g,
              unsigned* __restrict__ cnt, unsigned* __restrict__ gen) {
  extern __shared__ _Float16 smem[];
  _Float16* xbuf0 = smem;                  // 64 x XK (x_t parity 0)
  _Float16* xbuf1 = xbuf0 + 64 * XK;       // 64 x XK (x_t parity 1)
  _Float16* hbuf0 = xbuf1 + 64 * XK;       // 64 x HK
  _Float16* hbuf1 = hbuf0 + 64 * HK;       // 64 x HK
  _Float16* wih0  = hbuf1 + 64 * HK;       // 32 x XK  (tail: W_out 16 x HK)
  _Float16* whh0  = wih0  + 32 * XK;       // 32 x HK
  _Float16* wih1  = whh0  + 32 * HK;       // 32 x HK
  _Float16* whh1  = wih1  + 32 * HK;       // 32 x HK

  const int tid  = threadIdx.x;
  const int g    = blockIdx.x;
  const int n0   = g * NCOL;
  const int lane = tid & 31;
  const int wave = tid >> 5;
  const int mt   = wave >> 1;             // M tile 0..3
  const int nt   = wave & 1;              // N tile 0..1
  const int fr   = lane & 15;

  const unsigned hbuf0_off = lds_off(hbuf0);
  const unsigned hbuf1_off = lds_off(hbuf1);

  // ---- one-time: stage weight slices (fp32 -> f16) into LDS ----
  for (int i = tid; i < NCOL * DIN_; i += THREADS) {
    int r = i / DIN_, c = i % DIN_;
    wih0[r * XK + c] = (_Float16)Wih0[(n0 + r) * DIN_ + c];
  }
  for (int i = tid; i < NCOL * H_; i += THREADS) {
    int r = i / H_, c = i % H_;
    whh0[r * HK + c] = (_Float16)Whh0[(n0 + r) * H_ + c];
    wih1[r * HK + c] = (_Float16)Wih1[(n0 + r) * H_ + c];
    whh1[r * HK + c] = (_Float16)Whh1[(n0 + r) * H_ + c];
  }
  for (int i = tid; i < B_ * H_; i += THREADS) {
    int r = i / H_, c = i % H_;
    hbuf0[r * HK + c] = (_Float16)h0in[r * H_ + c];          // initial h0
    hbuf1[r * HK + c] = (_Float16)h0in[B_ * H_ + r * H_ + c];
  }

  const int   colg  = n0 + nt * 16 + fr;
  const float bias0 = bih0[colg] + bhh0[colg];
  const float bias1 = bih1[colg] + bhh1[colg];

  // stage x_0 into xbuf0 (each thread: one quarter-row, 64 values)
  const int xb = tid >> 2;
  const int xq = (tid & 3) * 64;
  {
    const float* src = x + (size_t)xb * T_ * DIN_ + xq;
    _Float16* dst = xbuf0 + xb * XK + xq;
#pragma unroll
    for (int c = 0; c < 64; c += 8) {
      float4 f0 = *(const float4*)(src + c);
      float4 f1 = *(const float4*)(src + c + 4);
      v8h hv;
      hv[0] = (_Float16)f0.x; hv[1] = (_Float16)f0.y;
      hv[2] = (_Float16)f0.z; hv[3] = (_Float16)f0.w;
      hv[4] = (_Float16)f1.x; hv[5] = (_Float16)f1.y;
      hv[6] = (_Float16)f1.z; hv[7] = (_Float16)f1.w;
      *(v8h*)(dst + c) = hv;
    }
  }
  __syncthreads();

  unsigned barnum = 0;
#pragma unroll 1
  for (int t = 0; t < T_; ++t) {
    const int p = t & 1;
    const _Float16* xb_cur = p ? xbuf1 : xbuf0;
    _Float16*       xb_nxt = p ? xbuf0 : xbuf1;

    // (b) layer0: h0' = tanh(x_t Wih0^T + h0 Whh0^T + bias0)   [slice 64x32]
    v8f acc0 = {0.f, 0.f, 0.f, 0.f, 0.f, 0.f, 0.f, 0.f};
    acc0 = gemm_acc(acc0, xb_cur, XK, mt * 16 + fr, wih0, XK, nt * 16 + fr, DIN_, lane);
    acc0 = gemm_acc(acc0, hbuf0,  HK, mt * 16 + fr, whh0, HK, nt * 16 + fr, H_,   lane);

    // (c) epilogue + publish slice to parity-p exchange buffer (f16)
    {
      const int row0 = mt * 16 + ((lane & 16) >> 1);
      _Float16* dst = h0g + (size_t)p * B_ * H_ + colg;
#pragma unroll
      for (int i = 0; i < 8; ++i)
        dst[(row0 + i) * H_] = (_Float16)tanhf(acc0[i] + bias0);
    }

    // (a+) stage x_{t+1} into the other parity buffer: overlaps barrier wait
    if (t + 1 < T_) {
      const float* src = x + ((size_t)xb * T_ + (t + 1)) * DIN_ + xq;
      _Float16* dst = xb_nxt + xb * XK + xq;
#pragma unroll
      for (int c = 0; c < 64; c += 8) {
        float4 f0 = *(const float4*)(src + c);
        float4 f1 = *(const float4*)(src + c + 4);
        v8h hv;
        hv[0] = (_Float16)f0.x; hv[1] = (_Float16)f0.y;
        hv[2] = (_Float16)f0.z; hv[3] = (_Float16)f0.w;
        hv[4] = (_Float16)f1.x; hv[5] = (_Float16)f1.y;
        hv[6] = (_Float16)f1.z; hv[7] = (_Float16)f1.w;
        *(v8h*)(dst + c) = hv;
      }
      if (t + 2 < T_) __builtin_prefetch(src + DIN_, 0, 3);
    }

    // (d) device barrier: h0[t] complete everywhere; h1[t-1] also complete
    grid_barrier(cnt, gen, barnum++);

    // (e) DMA full h0[t] and h1[t-1] into padded LDS via Tensor Data Mover
#if USE_TDM
    if (wave == 0) {
      tdm_load_2d_f16(h0g + (size_t)p * B_ * H_,       hbuf0_off, H_, B_, H_);
      tdm_load_2d_f16(h1g + (size_t)(1 - p) * B_ * H_, hbuf1_off, H_, B_, H_);
      __builtin_amdgcn_s_wait_tensorcnt(0);
    }
#else
    {
      const _Float16* s0 = h0g + (size_t)p * B_ * H_;
      const _Float16* s1 = h1g + (size_t)(1 - p) * B_ * H_;
      for (int i = tid; i < (B_ * H_) / 8; i += THREADS) {
        int r = i >> 6, c = (i & 63) * 8;
        *(v8h*)(hbuf0 + r * HK + c) = *(const v8h*)(s0 + r * H_ + c);
        *(v8h*)(hbuf1 + r * HK + c) = *(const v8h*)(s1 + r * H_ + c);
      }
    }
#endif
    __syncthreads();

    // (g) layer1: h1' = tanh(h0' Wih1^T + h1 Whh1^T + bias1)
    v8f acc1 = {0.f, 0.f, 0.f, 0.f, 0.f, 0.f, 0.f, 0.f};
    acc1 = gemm_acc(acc1, hbuf0, HK, mt * 16 + fr, wih1, HK, nt * 16 + fr, H_, lane);
    acc1 = gemm_acc(acc1, hbuf1, HK, mt * 16 + fr, whh1, HK, nt * 16 + fr, H_, lane);
    {
      const int row0 = mt * 16 + ((lane & 16) >> 1);
      _Float16* dst = h1g + (size_t)p * B_ * H_ + colg;
#pragma unroll
      for (int i = 0; i < 8; ++i)
        dst[(row0 + i) * H_] = (_Float16)tanhf(acc1[i] + bias1);
    }
    // hbuf0 now holds h0[t] -> A-operand for next step's layer0
  }

  // ---- tail: out = h1[T-1] @ Wout^T + bout  (WGs 0..7, 16 cols each) ----
  grid_barrier(cnt, gen, barnum++);
#if USE_TDM
  if (wave == 0) {
    tdm_load_2d_f16(h1g + (size_t)((T_ - 1) & 1) * B_ * H_, hbuf1_off, H_, B_, H_);
    __builtin_amdgcn_s_wait_tensorcnt(0);
  }
#else
  {
    const _Float16* s1 = h1g + (size_t)((T_ - 1) & 1) * B_ * H_;
    for (int i = tid; i < (B_ * H_) / 8; i += THREADS) {
      int r = i >> 6, c = (i & 63) * 8;
      *(v8h*)(hbuf1 + r * HK + c) = *(const v8h*)(s1 + r * H_ + c);
    }
  }
#endif
  if (g < DOUT_ / 16) {
    for (int i = tid; i < 16 * H_; i += THREADS) {   // reuse wih0 region
      int r = i / H_, c = i % H_;
      wih0[r * HK + c] = (_Float16)Wout[(g * 16 + r) * H_ + c];
    }
  }
  __syncthreads();
  if (g < DOUT_ / 16 && wave < 4) {
    v8f acc = {0.f, 0.f, 0.f, 0.f, 0.f, 0.f, 0.f, 0.f};
    acc = gemm_acc(acc, hbuf1, HK, wave * 16 + fr, wih0, HK, fr, H_, lane);
    const int row0 = wave * 16 + ((lane & 16) >> 1);
    const int oc   = g * 16 + fr;
    const float bo = bout[oc];
#pragma unroll
    for (int i = 0; i < 8; ++i)
      out[(size_t)(row0 + i) * DOUT_ + oc] = acc[i] + bo;
  }
}

// ---------------- host side --------------------------------------------------
extern "C" void kernel_launch(void* const* d_in, const int* in_sizes, int n_in,
                              void* d_out, int out_size, void* d_ws, size_t ws_size,
                              hipStream_t stream) {
  (void)in_sizes; (void)n_in; (void)out_size; (void)ws_size;
  const float* x    = (const float*)d_in[0];
  const float* h0in = (const float*)d_in[1];
  const float* Wih0 = (const float*)d_in[2];
  const float* bih0 = (const float*)d_in[3];
  const float* Whh0 = (const float*)d_in[4];
  const float* bhh0 = (const float*)d_in[5];
  const float* Wih1 = (const float*)d_in[6];
  const float* bih1 = (const float*)d_in[7];
  const float* Whh1 = (const float*)d_in[8];
  const float* bhh1 = (const float*)d_in[9];
  const float* Wout = (const float*)d_in[10];
  const float* bout = (const float*)d_in[11];
  float* out = (float*)d_out;

  _Float16* h0g = (_Float16*)d_ws;
  _Float16* h1g = h0g + 2 * B_ * H_;
  unsigned* cnt = (unsigned*)(h1g + 2 * B_ * H_);
  unsigned* gen = cnt + 1;

  hipFuncSetAttribute((const void*)rnn_scan,
                      hipFuncAttributeMaxDynamicSharedMemorySize, SMEM_BYTES);

  rnn_init<<<64, 256, 0, stream>>>(h0in, h0g, h1g, cnt, gen);
  rnn_scan<<<dim3(NWG), dim3(THREADS), SMEM_BYTES, stream>>>(
      x, h0in, Wih0, bih0, Whh0, bhh0, Wih1, bih1, Whh1, bhh1, Wout, bout,
      out, h0g, h1g, cnt, gen);
}
